// TransformerLayer_19164144075035
// MI455X (gfx1250) — compile-verified, matrix operand order checked by reading, not run on hardware
//
#include <hip/hip_runtime.h>
#include <cstdint>

typedef unsigned short u16;
typedef __attribute__((ext_vector_type(16))) __bf16 v16bf;
typedef __attribute__((ext_vector_type(8)))  float  v8f;

constexpr int Bsz = 8, S = 512, D = 1024, H = 16, DH = 64, DF = 4096;
constexpr int ROWS = Bsz * S;   // 4096

__device__ __forceinline__ u16 f2bf(float f) {
    uint32_t u = __float_as_uint(f);
    uint32_t r = u + 0x7FFFu + ((u >> 16) & 1u);   // round-to-nearest-even
    return (u16)(r >> 16);
}

union FragBF {                  // one wave32 16x32 bf16 A/B fragment (8 VGPRs)
    v16bf v;
    uint4 u4[2];
    u16   us[16];
};

// Async global -> LDS copy of 16 bytes per lane (ASYNCcnt-tracked).
__device__ __forceinline__ void async_b128(u16* lds, const u16* gptr) {
    uint32_t l = (uint32_t)(uintptr_t)lds;          // low 32 bits = LDS offset
    asm volatile("global_load_async_to_lds_b128 %0, %1, off"
                 :: "v"(l), "v"(gptr) : "memory");
}
__device__ __forceinline__ void wait_async0() {
    asm volatile("s_wait_asynccnt 0x0" ::: "memory");
}

// DPP butterfly reduction across a 16-lane row (lanes 0-15 / 16-31 separately).
template <int CTRL>
__device__ __forceinline__ float dpp_f(float x) {
    return __int_as_float(__builtin_amdgcn_mov_dpp(__float_as_int(x), CTRL, 0xf, 0xf, true));
}
__device__ __forceinline__ float red_max16(float v) {
    v = fmaxf(v, dpp_f<0xB1>(v));    // quad_perm [1,0,3,2]  (xor 1)
    v = fmaxf(v, dpp_f<0x4E>(v));    // quad_perm [2,3,0,1]  (xor 2)
    v = fmaxf(v, dpp_f<0x141>(v));   // row_half_mirror
    v = fmaxf(v, dpp_f<0x140>(v));   // row_mirror
    return v;
}
__device__ __forceinline__ float red_sum16(float v) {
    v += dpp_f<0xB1>(v);
    v += dpp_f<0x4E>(v);
    v += dpp_f<0x141>(v);
    v += dpp_f<0x140>(v);
    return v;
}

// ---------------------------------------------------------------------------
// Weight conversion: fp32 -> bf16, stored TRANSPOSED as W^T[N][K] so GEMM B
// tiles are row-major block copies (async-to-LDS friendly, no LDS scatter).
// ---------------------------------------------------------------------------
__global__ void k_conv_qkv(const float* __restrict__ Wq, const float* __restrict__ Wk,
                           const float* __restrict__ Wv, u16* __restrict__ dst) {
    int i = blockIdx.x * 256 + threadIdx.x;          // over 3*D*D, dst[n*D + d]
    int n = i / D, d = i % D;
    int which = n >> 10, nn = n & 1023;
    int h = nn >> 6, e = nn & 63;
    const float* W = (which == 0) ? Wq : (which == 1 ? Wk : Wv);
    dst[i] = f2bf(W[((size_t)h * D + d) * DH + e]);
}

__global__ void k_conv_t(const float* __restrict__ src, u16* __restrict__ dst, int N, int K) {
    int i = blockIdx.x * 256 + threadIdx.x;          // dst[n*K + k] = src[k*N + n]
    if (i < N * K) {
        int n = i / K, k = i % K;
        dst[i] = f2bf(src[(size_t)k * N + n]);
    }
}

// ---------------------------------------------------------------------------
// LayerNorm (block per row of D=1024, 256 threads) -> bf16
// ---------------------------------------------------------------------------
__global__ void k_layernorm(const float* __restrict__ x, const float* __restrict__ g,
                            const float* __restrict__ b, u16* __restrict__ out) {
    __shared__ float sb[256];
    const int row = blockIdx.x, tid = threadIdx.x;
    const float* xr = x + (size_t)row * D;
    float v0 = xr[tid], v1 = xr[tid + 256], v2 = xr[tid + 512], v3 = xr[tid + 768];
    sb[tid] = v0 + v1 + v2 + v3;
    __syncthreads();
    for (int o = 128; o > 0; o >>= 1) { if (tid < o) sb[tid] += sb[tid + o]; __syncthreads(); }
    float mean = sb[0] * (1.0f / D);
    __syncthreads();
    float d0 = v0 - mean, d1 = v1 - mean, d2 = v2 - mean, d3 = v3 - mean;
    sb[tid] = d0 * d0 + d1 * d1 + d2 * d2 + d3 * d3;
    __syncthreads();
    for (int o = 128; o > 0; o >>= 1) { if (tid < o) sb[tid] += sb[tid + o]; __syncthreads(); }
    float rstd = rsqrtf(sb[0] * (1.0f / D) + 1e-5f);
    u16* orow = out + (size_t)row * D;
    orow[tid]       = f2bf(d0 * rstd * g[tid]       + b[tid]);
    orow[tid + 256] = f2bf(d1 * rstd * g[tid + 256] + b[tid + 256]);
    orow[tid + 512] = f2bf(d2 * rstd * g[tid + 512] + b[tid + 512]);
    orow[tid + 768] = f2bf(d3 * rstd * g[tid + 768] + b[tid + 768]);
}

// ---------------------------------------------------------------------------
// bf16 WMMA GEMM: C[M][N] = A[M][K] * B[K][N] with B supplied TRANSPOSED
// (Bt[N][K]).  Block tile 128x128, K-step 64, double-buffered async-to-LDS
// staging, single barrier per K-step.  8 waves in 4(m) x 2(n); each wave:
// 32x64 = 2x4 WMMA tiles, 16 v_wmma_f32_16x16x32_bf16 per K-step.
// EPI 0: QKV permute -> bf16 (Q pre-scaled by 1/sqrt(DH), V stored transposed)
// EPI 1: +bias +res -> fp32      EPI 2: GELU(.+bias) -> bf16
// ---------------------------------------------------------------------------
constexpr int TSTRIDE = 72;               // 64 + 8 pad (u16)
constexpr int TSZ = 128 * TSTRIDE;        // one 128x64 tile in LDS (u16 count)

template <int EPI>
__global__ __launch_bounds__(256) void k_gemm(const u16* __restrict__ A, const u16* __restrict__ Bt,
                                              const float* __restrict__ bias, const float* __restrict__ res,
                                              float* __restrict__ outF, u16* __restrict__ outU,
                                              int N, int K) {
    __shared__ u16 smem[4 * TSZ];         // [buf 0/1][A tile | B tile]
    const int tid = threadIdx.x;
    const int lane = tid & 31, wave = tid >> 5;
    const int wm = wave & 3, wn = wave >> 2;           // 4(m) x 2(n)
    const int m0 = blockIdx.y * 128, n0 = blockIdx.x * 128;
    const int l15 = lane & 15, lhalf = lane >> 4, koff = lhalf * 8;
    const int ldr = tid >> 3, ldc = (tid & 7) * 8;     // loader: row 0..31 (+j*32), col chunk

    v8f acc[2][4] = {};

    auto issue_tile = [&](int k0, int buf) {
        u16* As = &smem[buf * 2 * TSZ];
        u16* Bs = As + TSZ;
#pragma unroll
        for (int j = 0; j < 4; j++) {
            int r = ldr + j * 32;
            async_b128(&As[r * TSTRIDE + ldc], &A[(size_t)(m0 + r) * K + k0 + ldc]);
            async_b128(&Bs[r * TSTRIDE + ldc], &Bt[(size_t)(n0 + r) * K + k0 + ldc]);
        }
    };

    const int nT = K / 64;
    issue_tile(0, 0);
    for (int t = 0; t < nT; t++) {
        wait_async0();
        __syncthreads();
        if (t + 1 < nT) issue_tile((t + 1) * 64, (t + 1) & 1);
        const u16* As = &smem[(t & 1) * 2 * TSZ];
        const u16* Bs = As + TSZ;
#pragma unroll
        for (int kc = 0; kc < 64; kc += 32) {
            FragBF a[2], b[4];
#pragma unroll
            for (int mt = 0; mt < 2; mt++) {
                const u16* p = &As[(wm * 32 + mt * 16 + l15) * TSTRIDE + kc + koff];
                a[mt].u4[0] = *(const uint4*)p;
                a[mt].u4[1] = *(const uint4*)(p + 16);
            }
#pragma unroll
            for (int nt = 0; nt < 4; nt++) {
                const u16* p = &Bs[(wn * 64 + nt * 16 + l15) * TSTRIDE + kc + koff];
                b[nt].u4[0] = *(const uint4*)p;
                b[nt].u4[1] = *(const uint4*)(p + 16);
            }
#pragma unroll
            for (int mt = 0; mt < 2; mt++)
#pragma unroll
                for (int nt = 0; nt < 4; nt++)
                    acc[mt][nt] = __builtin_amdgcn_wmma_f32_16x16x32_bf16(
                        false, a[mt].v, false, b[nt].v, (short)0, acc[mt][nt], false, false);
        }
    }

    constexpr size_t planeE = (size_t)Bsz * H * S * DH;
#pragma unroll
    for (int mt = 0; mt < 2; mt++) {
#pragma unroll
        for (int nt = 0; nt < 4; nt++) {
            int col = n0 + wn * 64 + nt * 16 + l15;
#pragma unroll
            for (int r = 0; r < 8; r++) {
                int row = m0 + wm * 32 + mt * 16 + r + 8 * lhalf;
                float vacc = acc[mt][nt][r];
                if constexpr (EPI == 0) {
                    int which = col >> 10;               // /D : 0=Q 1=K 2=V
                    int nn = col & 1023;
                    int h = nn >> 6, e = nn & 63;
                    int bb = row >> 9, ss = row & 511;   // /S, %S
                    if (which == 0) vacc *= 0.125f;      // fold 1/sqrt(DH) into Q
                    size_t idx;
                    if (which == 2)                       // V stored transposed [b][h][dh][s]
                        idx = 2 * planeE + (((size_t)bb * H + h) * DH + e) * S + ss;
                    else
                        idx = (size_t)which * planeE + (((size_t)bb * H + h) * S + ss) * DH + e;
                    outU[idx] = f2bf(vacc);
                } else if constexpr (EPI == 1) {
                    size_t idx = (size_t)row * N + col;
                    outF[idx] = vacc + bias[col] + res[idx];
                } else {
                    float tt = vacc + bias[col];
                    float gl = 0.5f * tt * (1.0f + erff(tt * 0.70710678118654752f));
                    outU[(size_t)row * N + col] = f2bf(gl);
                }
            }
        }
    }
}

// ---------------------------------------------------------------------------
// Causal flash attention per (b,h).  Block = 8 waves = 128 query rows; each
// wave owns 16 rows.  32-key tiles, double-buffered async-to-LDS K/V staging
// (V pre-transposed in global), DPP softmax reductions, 8 full WMMAs/tile.
// ---------------------------------------------------------------------------
__global__ __launch_bounds__(256) void k_attention(const u16* __restrict__ qkv, u16* __restrict__ ctx) {
    __shared__ u16 Ks[2][32][72];    // [buf][key t][dh]   (+8 pad)
    __shared__ u16 Vs[2][64][40];    // [buf][dh][key t]   (+8 pad)
    __shared__ u16 Ps[8][16][40];    // per-wave P tile (16 q x 32 keys)
    const int tid = threadIdx.x;
    const int lane = tid & 31, wave = tid >> 5;
    const int l15 = lane & 15, lhalf = lane >> 4, koff = lhalf * 8;
    const int bh = blockIdx.x;                    // b*H + h
    const int qBase = blockIdx.y * 128;
    const int q0w = qBase + wave * 16;
    const size_t plane = (size_t)Bsz * H * S * DH;
    const u16* qp = qkv + (size_t)bh * S * DH;
    const u16* kp = qkv + plane + (size_t)bh * S * DH;
    const u16* vp = qkv + 2 * plane + (size_t)bh * DH * S;   // transposed [dh][s]

    // Q fragments (already scaled by 1/sqrt(DH)) for this wave's 16 rows.
    FragBF qa0, qa1;
    {
        const u16* qrow = qp + (size_t)(q0w + l15) * DH;
        qa0.u4[0] = *(const uint4*)&qrow[koff];
        qa0.u4[1] = *(const uint4*)&qrow[koff + 16];
        qa1.u4[0] = *(const uint4*)&qrow[32 + koff];
        qa1.u4[1] = *(const uint4*)&qrow[32 + koff + 16];
    }

    auto issue_kv = [&](int kt, int buf) {
        int t = tid >> 3, c8 = (tid & 7) * 8;              // K: 32 keys x 64 dh
        async_b128(&Ks[buf][t][c8], &kp[(size_t)(kt * 32 + t) * DH + c8]);
        int dh = tid >> 2, c = (tid & 3) * 8;              // V^T: 64 dh x 32 keys
        async_b128(&Vs[buf][dh][c], &vp[(size_t)dh * S + kt * 32 + c]);
    };

    v8f ov[4] = {};                 // O accum: 16 rows x 64 dh
    float mrow[8], lrow[8];
#pragma unroll
    for (int r = 0; r < 8; r++) { mrow[r] = -3.0e38f; lrow[r] = 0.f; }

    const int ktEnd = (qBase + 127) >> 5;         // 32-key tiles
    issue_kv(0, 0);
    for (int kt = 0; kt <= ktEnd; kt++) {
        wait_async0();
        __syncthreads();
        if (kt + 1 <= ktEnd) issue_kv(kt + 1, (kt + 1) & 1);
        const int cur = kt & 1;

        // scores: Q(16x64) @ K^T(64x32) -> two 16x16 C tiles, 4 full WMMAs
        FragBF kb;
        v8f sc0 = {}, sc1 = {};
        kb.u4[0] = *(const uint4*)&Ks[cur][l15][koff];
        kb.u4[1] = *(const uint4*)&Ks[cur][l15][koff + 16];
        sc0 = __builtin_amdgcn_wmma_f32_16x16x32_bf16(false, qa0.v, false, kb.v, (short)0, sc0, false, false);
        kb.u4[0] = *(const uint4*)&Ks[cur][l15][32 + koff];
        kb.u4[1] = *(const uint4*)&Ks[cur][l15][32 + koff + 16];
        sc0 = __builtin_amdgcn_wmma_f32_16x16x32_bf16(false, qa1.v, false, kb.v, (short)0, sc0, false, false);
        kb.u4[0] = *(const uint4*)&Ks[cur][16 + l15][koff];
        kb.u4[1] = *(const uint4*)&Ks[cur][16 + l15][koff + 16];
        sc1 = __builtin_amdgcn_wmma_f32_16x16x32_bf16(false, qa0.v, false, kb.v, (short)0, sc1, false, false);
        kb.u4[0] = *(const uint4*)&Ks[cur][16 + l15][32 + koff];
        kb.u4[1] = *(const uint4*)&Ks[cur][16 + l15][32 + koff + 16];
        sc1 = __builtin_amdgcn_wmma_f32_16x16x32_bf16(false, qa1.v, false, kb.v, (short)0, sc1, false, false);

        const int colg0 = kt * 32 + l15, colg1 = colg0 + 16;
        float p0v[8], p1v[8], fac[8];
#pragma unroll
        for (int r = 0; r < 8; r++) {
            int rowg = q0w + r + 8 * lhalf;
            float s0 = sc0[r];                              // scale already folded into Q
            float s1 = sc1[r];
            if (colg0 > rowg) s0 = -3.0e38f;                // causal mask
            if (colg1 > rowg) s1 = -3.0e38f;
            float tm = red_max16(fmaxf(s0, s1));            // DPP row reduction
            float mnew = fmaxf(mrow[r], tm);
            float f = expf(mrow[r] - mnew);
            float p0 = expf(s0 - mnew);
            float p1 = expf(s1 - mnew);
            if (colg0 > rowg) p0 = 0.f;
            if (colg1 > rowg) p1 = 0.f;
            float ps = red_sum16(p0 + p1);                  // DPP row reduction
            lrow[r] = lrow[r] * f + ps;
            mrow[r] = mnew;
            fac[r] = f;
            p0v[r] = p0;
            p1v[r] = p1;
        }

        // P tile -> per-wave LDS (C-frag layout -> A-frag layout); same-wave LDS in-order.
#pragma unroll
        for (int r = 0; r < 8; r++) {
            Ps[wave][r + 8 * lhalf][l15]      = f2bf(p0v[r]);
            Ps[wave][r + 8 * lhalf][16 + l15] = f2bf(p1v[r]);
        }
        asm volatile("s_wait_dscnt 0x0" ::: "memory");

#pragma unroll
        for (int c = 0; c < 4; c++)
#pragma unroll
            for (int r = 0; r < 8; r++) ov[c][r] *= fac[r];

        FragBF pa;                                          // full 16x32 A fragment
        pa.u4[0] = *(const uint4*)&Ps[wave][l15][koff];
        pa.u4[1] = *(const uint4*)&Ps[wave][l15][koff + 16];
#pragma unroll
        for (int c = 0; c < 4; c++) {
            FragBF vb;                                      // full 32x16 B fragment
            vb.u4[0] = *(const uint4*)&Vs[cur][c * 16 + l15][koff];
            vb.u4[1] = *(const uint4*)&Vs[cur][c * 16 + l15][koff + 16];
            ov[c] = __builtin_amdgcn_wmma_f32_16x16x32_bf16(false, pa.v, false, vb.v, (short)0, ov[c], false, false);
        }
    }

    // normalize and write ctx[b][s][h*DH + e] as bf16
    const int b = bh >> 4, h = bh & 15;
#pragma unroll
    for (int r = 0; r < 8; r++) {
        float inv = 1.0f / lrow[r];
        int srow = q0w + r + 8 * lhalf;
        size_t base = ((size_t)(b * S + srow)) * D + h * DH;
#pragma unroll
        for (int c = 0; c < 4; c++)
            ctx[base + c * 16 + l15] = f2bf(ov[c][r] * inv);
    }
}

// ---------------------------------------------------------------------------
extern "C" void kernel_launch(void* const* d_in, const int* in_sizes, int n_in,
                              void* d_out, int out_size, void* d_ws, size_t ws_size,
                              hipStream_t stream) {
    (void)in_sizes; (void)n_in; (void)out_size; (void)ws_size;
    const float* x   = (const float*)d_in[0];
    const float* g1  = (const float*)d_in[1];
    const float* be1 = (const float*)d_in[2];
    const float* Wq  = (const float*)d_in[3];
    const float* Wk  = (const float*)d_in[4];
    const float* Wv  = (const float*)d_in[5];
    const float* Wo  = (const float*)d_in[6];
    const float* bo  = (const float*)d_in[7];
    const float* g2  = (const float*)d_in[8];
    const float* be2 = (const float*)d_in[9];
    const float* W1  = (const float*)d_in[10];
    const float* fb1 = (const float*)d_in[11];
    const float* W2  = (const float*)d_in[12];
    const float* fb2 = (const float*)d_in[13];
    float* out = (float*)d_out;

    char* ws = (char*)d_ws;
    auto alloc = [&](size_t bytes) -> char* {
        char* p = ws; ws += (bytes + 255) & ~(size_t)255; return p;
    };
    u16*   wqkvT = (u16*)alloc((size_t)3 * D * D * 2);   // [3D][D]
    u16*   woT   = (u16*)alloc((size_t)D * D * 2);       // [D][D]
    u16*   w1T   = (u16*)alloc((size_t)DF * D * 2);      // [DF][D]
    u16*   w2T   = (u16*)alloc((size_t)D * DF * 2);      // [D][DF]
    u16*   h1    = (u16*)alloc((size_t)ROWS * D * 2);
    u16*   qkv   = (u16*)alloc((size_t)3 * ROWS * D * 2);
    u16*   ctx   = (u16*)alloc((size_t)ROWS * D * 2);
    float* xmid  = (float*)alloc((size_t)ROWS * D * 4);
    u16*   h2    = (u16*)alloc((size_t)ROWS * D * 2);
    u16*   ffn1  = (u16*)alloc((size_t)ROWS * DF * 2);

    dim3 blk(256);
    k_conv_qkv<<<(3 * D * D) / 256, blk, 0, stream>>>(Wq, Wk, Wv, wqkvT);
    k_conv_t<<<(D * D) / 256, blk, 0, stream>>>(Wo, woT, D, D);
    k_conv_t<<<(DF * D) / 256, blk, 0, stream>>>(W1, w1T, DF, D);
    k_conv_t<<<(D * DF) / 256, blk, 0, stream>>>(W2, w2T, D, DF);

    k_layernorm<<<ROWS, blk, 0, stream>>>(x, g1, be1, h1);
    k_gemm<0><<<dim3(3 * D / 128, ROWS / 128), blk, 0, stream>>>(h1, wqkvT, nullptr, nullptr, nullptr, qkv, 3 * D, D);
    k_attention<<<dim3(Bsz * H, S / 128), blk, 0, stream>>>(qkv, ctx);
    k_gemm<1><<<dim3(D / 128, ROWS / 128), blk, 0, stream>>>(ctx, woT, bo, x, xmid, nullptr, D, D);
    k_layernorm<<<ROWS, blk, 0, stream>>>(xmid, g2, be2, h2);
    k_gemm<2><<<dim3(DF / 128, ROWS / 128), blk, 0, stream>>>(h2, w1T, fb1, nullptr, nullptr, ffn1, DF, D);
    k_gemm<1><<<dim3(D / 128, ROWS / 128), blk, 0, stream>>>(ffn1, w2T, fb2, xmid, out, nullptr, D, DF);
}
